// LSTMModelStates_6098853560815
// MI455X (gfx1250) — matrix operand order repaired and checked by reading, not compiled, not run on hardware
//
#include <hip/hip_runtime.h>
#include <cstdint>
#include <cstddef>

// Problem constants (from reference): B=128, T=365, D=32, H=512
#define B_    128
#define T_    365
#define D_    32
#define H_    512
#define G4H   2048          // 4*H
#define KTOT  544           // D + H (fused [x_t, h] @ [Wx; Wh])
#define KK    17            // KTOT / 32 k-steps per timestep
#define NT    128           // G4H / 16 column tiles
#define PITCH 520           // LDS h row pitch in bf16 (1040B rows: 16B-aligned, bank-spread)

typedef __attribute__((ext_vector_type(16))) __bf16 v16bf;
typedef __attribute__((ext_vector_type(8)))  __bf16 v8bf;
typedef __attribute__((ext_vector_type(8)))  float  v8f;
typedef __attribute__((ext_vector_type(4)))  float  v4f;

__device__ __forceinline__ float sigmoidf_(float z) {
  return 1.0f / (1.0f + __expf(-z));
}

// ---------------------------------------------------------------------------
// Prep: pack [Wx; Wh] (fp32, row-major KxN, K=544, N=2048) into bf16 B-matrix
// fragments for v_wmma_f32_16x16x32_bf16. Fragment (kk, nt) = 32x16 tile,
// stored as 32 lanes x 16 bf16 (32B contiguous per lane):
//   lane l, element e  ->  K = kk*32 + (e/8)*16 + (l/16)*8 + (e%8),  N = nt*16 + (l%16)
// ---------------------------------------------------------------------------
__global__ void prep_wfrag(const float* __restrict__ Wx,
                           const float* __restrict__ Wh,
                           __bf16* __restrict__ wfrag) {
  int id = blockIdx.x * blockDim.x + threadIdx.x;
  const int total = KK * NT * 512;
  if (id >= total) return;
  int frag = id >> 9;            // /512
  int r    = id & 511;
  int lane = r >> 4;
  int e    = r & 15;
  int kk   = frag / NT;
  int nt   = frag - kk * NT;
  int K = kk * 32 + ((e >> 3) << 4) + ((lane >> 4) << 3) + (e & 7);
  int N = nt * 16 + (lane & 15);
  float v = (K < D_) ? Wx[(size_t)K * G4H + N]
                     : Wh[(size_t)(K - D_) * G4H + N];
  wfrag[id] = (__bf16)v;
}

// ---------------------------------------------------------------------------
// Persistent LSTM kernel. grid.x = 8 (batch tiles of 16 rows), block = 512
// threads = 16 waves (wave32). Wave wv owns h-columns [wv*32, wv*32+32) in
// ALL four gates (8 accumulator tiles), so the cell update is register-local.
// h state: bf16 in LDS (A operands via ds_load_b128). c state: fp32 regs.
// Per step per wave: 17 k-iters x 8 tiles = 136 v_wmma_f32_16x16x32_bf16.
// ---------------------------------------------------------------------------
__global__ __launch_bounds__(512)
void lstm_persistent(const float* __restrict__ x,      // [B,T,D] fp32
                     const float* __restrict__ bias,   // [4H]    fp32
                     const __bf16* __restrict__ wfrag, // bf16 B fragments
                     float* __restrict__ out) {        // [B,T,H] fp32
  __shared__ __bf16 hlds[16 * PITCH];

  const int tid   = threadIdx.x;
  const int lane  = tid & 31;
  const int wv    = tid >> 5;          // 0..15
  const int b0    = blockIdx.x * 16;   // batch tile base
  const int nlo   = lane & 15;
  const int khalf = lane >> 4;         // 0 or 1
  const int arow  = nlo;               // A-fragment row this lane feeds

  // zero initial h
  for (int i = tid; i < 16 * PITCH; i += 512) hlds[i] = (__bf16)0.0f;

  // cell state (C-fragment layout): tile s=0..1, acc vgpr r=0..7
  float creg[2][8];
#pragma unroll
  for (int s = 0; s < 2; ++s)
#pragma unroll
    for (int r = 0; r < 8; ++r) creg[s][r] = 0.0f;

  // bias per (gate, subtile): depends only on N = lane%16 within the tile
  float bias_v[4][2];
#pragma unroll
  for (int g = 0; g < 4; ++g)
#pragma unroll
    for (int s = 0; s < 2; ++s)
      bias_v[g][s] = bias[g * H_ + wv * 32 + s * 16 + nlo];

  __syncthreads();

#pragma unroll 1
  for (int t = 0; t < T_; ++t) {
    // Launder a zero SGPR offset (not the pointer) each time step: defeats
    // LICM of the t-invariant weight loads (which previously spilled 4.3KB
    // per lane to scratch) while PRESERVING the global addrspace provenance
    // of wfrag -> global_load_b128 (LOADcnt only), not flat_load (which
    // would also tick DScnt and falsely order against the LDS A loads).
    size_t toff = 0;
    asm volatile("" : "+s"(toff));
    const __bf16* wf = wfrag + toff;

    // ---- A fragment for kk=0: x[b0+arow, t, 0:32] -> bf16, A 16x32 layout ----
    v16bf a0;
    {
      const float* xr = x + ((size_t)(b0 + arow) * T_ + t) * D_;
      const int c0 = khalf * 8;
      v4f p0 = *(const v4f*)(xr + c0);
      v4f p1 = *(const v4f*)(xr + c0 + 4);
      v4f p2 = *(const v4f*)(xr + 16 + c0);
      v4f p3 = *(const v4f*)(xr + 16 + c0 + 4);
#pragma unroll
      for (int i = 0; i < 4; ++i) {
        a0[i]      = (__bf16)p0[i];
        a0[4 + i]  = (__bf16)p1[i];
        a0[8 + i]  = (__bf16)p2[i];
        a0[12 + i] = (__bf16)p3[i];
      }
    }

    // ---- init accumulators with bias ----
    v8f acc[4][2];
#pragma unroll
    for (int g = 0; g < 4; ++g)
#pragma unroll
      for (int s = 0; s < 2; ++s) {
        v8f az;
#pragma unroll
        for (int r = 0; r < 8; ++r) az[r] = bias_v[g][s];
        acc[g][s] = az;
      }

    // ---- kk = 0 : x_t @ Wx  (issue all 8 fragment loads, then 8 WMMAs) ----
    {
      v16bf bfr[8];
#pragma unroll
      for (int u = 0; u < 8; ++u) {
        const int nt = (u >> 1) * 32 + wv * 2 + (u & 1);   // gate=u/2, s=u&1
        bfr[u] = *(const v16bf*)(wf + (size_t)nt * 512 + lane * 16);
      }
#pragma unroll
      for (int u = 0; u < 8; ++u)
        acc[u >> 1][u & 1] = __builtin_amdgcn_wmma_f32_16x16x32_bf16(
            false, a0, false, bfr[u], (short)0, acc[u >> 1][u & 1], false, false);
    }

    // ---- kk = 1..16 : h @ Wh (A from LDS bf16 h) ----
#pragma unroll 1
    for (int kk = 1; kk < KK; ++kk) {
      const int hb = (kk - 1) * 32 + khalf * 8;   // bf16 element offset
      union { v16bf v; v8bf h[2]; } au;
      au.h[0] = *(const v8bf*)&hlds[arow * PITCH + hb];
      au.h[1] = *(const v8bf*)&hlds[arow * PITCH + hb + 16];
      const v16bf a = au.v;

      const __bf16* wk = wf + (size_t)kk * (NT * 512);
      v16bf bfr[8];
#pragma unroll
      for (int u = 0; u < 8; ++u) {
        const int nt = (u >> 1) * 32 + wv * 2 + (u & 1);
        bfr[u] = *(const v16bf*)(wk + (size_t)nt * 512 + lane * 16);
      }
      if (kk + 1 < KK) {
        // hint next k-slice of the weight fragments (emits global_prefetch)
        __builtin_prefetch(wk + (size_t)NT * 512 + (size_t)(wv * 2) * 512, 0, 0);
      }
#pragma unroll
      for (int u = 0; u < 8; ++u)
        acc[u >> 1][u & 1] = __builtin_amdgcn_wmma_f32_16x16x32_bf16(
            false, a, false, bfr[u], (short)0, acc[u >> 1][u & 1], false, false);
    }

    // ---- gates + cell update (fully register-local) ----
    float hval[2][8];
#pragma unroll
    for (int s = 0; s < 2; ++s)
#pragma unroll
      for (int r = 0; r < 8; ++r) {
        const float zi = acc[0][s][r];
        const float zf = acc[1][s][r];
        const float zg = acc[2][s][r];
        const float zo = acc[3][s][r];
        const float ig = sigmoidf_(zi);
        const float fg = sigmoidf_(zf);
        const float gg = tanhf(zg);
        const float og = sigmoidf_(zo);
        const float cn = fg * creg[s][r] + ig * gg;
        creg[s][r] = cn;
        hval[s][r] = og * tanhf(cn);
      }

    __syncthreads();   // everyone finished reading old h from LDS

    // ---- publish new h (bf16 -> LDS) and stream fp32 h to output ----
    const int mrow0 = khalf * 8;   // C layout: M = r + 8*(lane/16), N = lane%16
#pragma unroll
    for (int s = 0; s < 2; ++s) {
      const int hc = wv * 32 + s * 16 + nlo;
#pragma unroll
      for (int r = 0; r < 8; ++r) {
        const int M = mrow0 + r;
        const float hv = hval[s][r];
        hlds[M * PITCH + hc] = (__bf16)hv;
        out[((size_t)(b0 + M) * T_ + t) * H_ + hc] = hv;
      }
    }
    __syncthreads();   // new h visible before next step's A loads
  }
}

// ---------------------------------------------------------------------------
// Inputs (setup_inputs order): x [B,T,D] f32, Wx [D,4H] f32, Wh [H,4H] f32,
// b [4H] f32. Output: h sequence [B,T,H] f32. d_ws: 2.13 MB bf16 fragments.
// ---------------------------------------------------------------------------
extern "C" void kernel_launch(void* const* d_in, const int* in_sizes, int n_in,
                              void* d_out, int out_size, void* d_ws, size_t ws_size,
                              hipStream_t stream) {
  (void)in_sizes; (void)n_in; (void)out_size; (void)ws_size;
  const float* x  = (const float*)d_in[0];
  const float* Wx = (const float*)d_in[1];
  const float* Wh = (const float*)d_in[2];
  const float* b  = (const float*)d_in[3];
  float* out = (float*)d_out;
  __bf16* wfrag = (__bf16*)d_ws;   // KK*NT*512 bf16 = 2,228,224 B

  const int total = KK * NT * 512;
  prep_wfrag<<<(total + 255) / 256, 256, 0, stream>>>(Wx, Wh, wfrag);
  lstm_persistent<<<B_ / 16, 512, 0, stream>>>(x, b, wfrag, out);
}
